// GraphAttention_18726057411373
// MI455X (gfx1250) — matrix operand order repaired and checked by reading, not compiled, not run on hardware
//
#include <hip/hip_runtime.h>
#include <hip/hip_bf16.h>
#include <math.h>

typedef __attribute__((ext_vector_type(16))) _Float16 v16h;
typedef __attribute__((ext_vector_type(8)))  _Float16 v8h;
typedef __attribute__((ext_vector_type(8)))  float    v8f;

#define NS 64
#define NV 16
#define NHEAD 8
#define RBF 128
#define MID 64
#define NW 176
#define ALPHA_C 10
#define VEC_C 12
#define D_IN 112
#define D_OUT 112
#define AGG_C 368            // NHEAD * (ALPHA_C + 3*VEC_C) = 8*46
#define INV_S3 0.5773502691896258f
#define INV_S2 0.7071067811865475f
#define NFRAG1 16            // layer-1 B fragments: 4 K-chunks x 4 N-tiles
#define NFRAG2 22            // layer-2 B fragments: 2 K-chunks x 11 N-tiles
#define NFRAG  (NFRAG1 + NFRAG2)

// ---------- monotonic float <-> uint encoding for atomic segment-max ----------
__device__ __forceinline__ unsigned fenc(float x) {
  unsigned u = __float_as_uint(x);
  return (u & 0x80000000u) ? ~u : (u | 0x80000000u);
}
__device__ __forceinline__ float fdec(unsigned e) {
  return (e & 0x80000000u) ? __uint_as_float(e & 0x7fffffffu)
                           : __uint_as_float(~e);
}

// ============================ init ============================
__global__ void k_init(unsigned* __restrict__ amax_enc, float* __restrict__ den,
                       float* __restrict__ agg, int N) {
  long idx = (long)blockIdx.x * blockDim.x + threadIdx.x;
  long n8 = (long)N * 8;
  long tot = n8 + n8 + (long)N * AGG_C;
  if (idx >= tot) return;
  if (idx < n8)            amax_enc[idx] = 0x007FFFFFu;     // enc(-inf)
  else if (idx < 2 * n8)   den[idx - n8] = 0.f;
  else                     agg[idx - 2 * n8] = 0.f;
}

// ====== pack rad_w1/rad_w2 into f16 WMMA B-fragment layout (one-shot, 19456 elems) ======
// fragment = 32 lanes x 16 halves, lane-major (32 B per lane, coalesced per wave)
__global__ void k_pack(const float* __restrict__ w1, const float* __restrict__ w2,
                       _Float16* __restrict__ wp) {
  int idx = blockIdx.x * blockDim.x + threadIdx.x;
  if (idx >= NFRAG * 512) return;
  const int frag = idx >> 9;
  const int r    = idx & 511;
  const int lane = r >> 4;
  const int e    = r & 15;
  const int g    = lane >> 4;
  const int row  = lane & 15;
  float val;
  if (frag < NFRAG1) {
    const int k = frag >> 2, nb = frag & 3;
    const int K = k * 32 + g * 8 + e + (e >= 8 ? 8 : 0); // B-operand K map
    val = w1[K * MID + nb * 16 + row];
  } else {
    const int f = frag - NFRAG1;
    const int k = f / 11, nb = f % 11;
    const int K = k * 32 + g * 8 + e + (e >= 8 ? 8 : 0);
    val = w2[K * NW + nb * 16 + row];
  }
  wp[idx] = (_Float16)val;
}

// ============================ node-side transforms ============================
__global__ void __launch_bounds__(256)
k_node(const float* __restrict__ x,
       const float* __restrict__ wss_g, const float* __restrict__ wsv_g,
       const float* __restrict__ wds_g, const float* __restrict__ wdv_g,
       float* __restrict__ ssrc, float* __restrict__ sdst,
       float* __restrict__ vsrc, float* __restrict__ vdst, int N) {
  __shared__ float wss[64 * 64], wds[64 * 64], wsv[16 * 16], wdv[16 * 16];
  for (int i = threadIdx.x; i < 64 * 64; i += 256) { wss[i] = wss_g[i]; wds[i] = wds_g[i]; }
  for (int i = threadIdx.x; i < 16 * 16; i += 256) { wsv[i] = wsv_g[i]; wdv[i] = wdv_g[i]; }
  __syncthreads();
  int nbase = blockIdx.x * 32;
  for (int i = threadIdx.x; i < 32 * 224; i += 256) {
    int ln = i / 224, c = i % 224;
    int n = nbase + ln;
    if (n >= N) continue;
    const float* row = x + (size_t)n * D_IN;
    if (c < 64) {
      float acc = 0.f;
#pragma unroll 8
      for (int k = 0; k < 64; ++k) acc += row[k] * wss[k * 64 + c];
      ssrc[(size_t)n * 64 + c] = acc;
    } else if (c < 128) {
      int cc = c - 64; float acc = 0.f;
#pragma unroll 8
      for (int k = 0; k < 64; ++k) acc += row[k] * wds[k * 64 + cc];
      sdst[(size_t)n * 64 + cc] = acc;
    } else if (c < 176) {
      int j = c - 128, e = j / 3, d = j % 3; float acc = 0.f;
#pragma unroll
      for (int k = 0; k < 16; ++k) acc += row[64 + k * 3 + d] * wsv[k * 16 + e];
      vsrc[(size_t)n * 48 + j] = acc;
    } else {
      int j = c - 176, e = j / 3, d = j % 3; float acc = 0.f;
#pragma unroll
      for (int k = 0; k < 16; ++k) acc += row[64 + k * 3 + d] * wdv[k * 16 + e];
      vdst[(size_t)n * 48 + j] = acc;
    }
  }
}

// ============================ WMMA radial MLP (16 edges / wave) ============================
struct WaveScratch {
  float    wbuf[16 * NW];    // layer-2 output (first 16*64 floats doubles as layer-1 preact)
  _Float16 hbuf[16 * MID];   // LN+SiLU activations, f16, A-operand staging
};

__device__ __forceinline__ void radial_mlp(WaveScratch& S, int ebase, int lane, int E,
                                           const float* __restrict__ xs,
                                           const _Float16* __restrict__ wpack,
                                           const float* __restrict__ b1,
                                           const float* __restrict__ b2) {
  const int row = lane & 15;   // M index of this lane
  const int g   = lane >> 4;   // lane-group (K sub-block select)
  int er = ebase + row;
  if (er >= E) er = E - 1;     // clamp: keep EXEC all-ones for WMMA
  const float*  xrow = xs + (size_t)er * RBF;
  const float4* x4   = (const float4*)xrow;
  const v16h*   bw   = (const v16h*)wpack;   // fragment f: bw[f*32 + lane]

  // ---- layer 1: (16 x 128) @ (128 x 64), K in 4 chunks of 32 ----
  // A-operand element e sits at K = kb + (e<8?0:16) + 8g + (e&7): two 8-float runs
  v16h a[4];
#pragma unroll
  for (int k = 0; k < 4; ++k) {
    const int base = (k * 32 + 8 * g) >> 2;       // in float4 units
    float4 q0 = x4[base + 0];
    float4 q1 = x4[base + 1];
    float4 q2 = x4[base + 4];                     // +16 floats
    float4 q3 = x4[base + 5];
    a[k] = (v16h){(_Float16)q0.x, (_Float16)q0.y, (_Float16)q0.z, (_Float16)q0.w,
                  (_Float16)q1.x, (_Float16)q1.y, (_Float16)q1.z, (_Float16)q1.w,
                  (_Float16)q2.x, (_Float16)q2.y, (_Float16)q2.z, (_Float16)q2.w,
                  (_Float16)q3.x, (_Float16)q3.y, (_Float16)q3.z, (_Float16)q3.w};
  }
  float* pre = S.wbuf;  // 16 x 64 overlay
#pragma unroll
  for (int nb = 0; nb < 4; ++nb) {
    const int col = nb * 16 + row;
    v8f c = {};
#pragma unroll
    for (int k = 0; k < 4; ++k) {
      v16h b = bw[(k * 4 + nb) * 32 + lane];      // pre-packed f16 fragment
      c = __builtin_amdgcn_wmma_f32_16x16x32_f16(false, a[k], false, b,
                                                 (short)0, c, false, false);
    }
    const float bias = b1[col];
#pragma unroll
    for (int r = 0; r < 8; ++r) pre[(r + 8 * g) * MID + col] = c[r] + bias;
  }
  __syncthreads();

  // ---- LayerNorm + SiLU, emit f16 activations ----
  if (lane < 16) {
    const float* p = pre + row * MID;
    float m = 0.f;
#pragma unroll 8
    for (int c2 = 0; c2 < MID; ++c2) m += p[c2];
    m *= (1.f / MID);
    float v = 0.f;
#pragma unroll 8
    for (int c2 = 0; c2 < MID; ++c2) { float d = p[c2] - m; v += d * d; }
    v *= (1.f / MID);
    const float rs = rsqrtf(v + 1e-5f);
#pragma unroll 8
    for (int c2 = 0; c2 < MID; ++c2) {
      float xn = (p[c2] - m) * rs;
      float sl = xn / (1.f + expf(-xn));                    // SiLU
      S.hbuf[row * MID + c2] = (_Float16)sl;
    }
  }
  __syncthreads();

  // ---- layer 2: (16 x 64) @ (64 x 176), K in 2 chunks of 32 ----
  v16h ha[2];
  {
    const v8h* hb = (const v8h*)(S.hbuf + row * MID);  // 8-half (16 B) runs, aligned
#pragma unroll
    for (int k = 0; k < 2; ++k) {
      v8h r0 = hb[(k * 32 + 8 * g) >> 3];
      v8h r1 = hb[((k * 32 + 16 + 8 * g) >> 3)];
      ha[k] = (v16h){r0[0], r0[1], r0[2], r0[3], r0[4], r0[5], r0[6], r0[7],
                     r1[0], r1[1], r1[2], r1[3], r1[4], r1[5], r1[6], r1[7]};
    }
  }
#pragma unroll
  for (int nb = 0; nb < 11; ++nb) {
    const int col = nb * 16 + row;
    v8f c = {};
#pragma unroll
    for (int k = 0; k < 2; ++k) {
      v16h b = bw[(NFRAG1 + k * 11 + nb) * 32 + lane];
      c = __builtin_amdgcn_wmma_f32_16x16x32_f16(false, ha[k], false, b,
                                                 (short)0, c, false, false);
    }
    const float bias = b2[col];
#pragma unroll
    for (int r = 0; r < 8; ++r) S.wbuf[(r + 8 * g) * NW + col] = c[r] + bias;
  }
}

// ============================ edge pass 1: attention logits ============================
__global__ void __launch_bounds__(128)
k_edge_alpha(const float* __restrict__ xs, const _Float16* __restrict__ wpack,
             const float* __restrict__ b1, const float* __restrict__ b2,
             const int* __restrict__ esrc, const int* __restrict__ edst,
             const float* __restrict__ eattr,
             const float* __restrict__ ssrc, const float* __restrict__ sdst,
             const float* __restrict__ vsrc, const float* __restrict__ vdst,
             const float* __restrict__ alpha_dot,
             float* __restrict__ alpha_out, unsigned* __restrict__ amax_enc, int E) {
  __shared__ WaveScratch SW[4];
  const int wave = threadIdx.x >> 5, lane = threadIdx.x & 31;
  radial_mlp(SW[wave], blockIdx.x * 64 + wave * 16, lane, E, xs, wpack, b1, b2);
  __syncthreads();

  const int el = threadIdx.x >> 1;
  const int e  = blockIdx.x * 64 + el;
  if (e >= E) return;
  const float* w = SW[el >> 4].wbuf + (el & 15) * NW;
  const int src = esrc[e], dst = edst[e];
  const float e0  = eattr[(size_t)e * 4 + 0];
  const float e1x = eattr[(size_t)e * 4 + 1];
  const float e1y = eattr[(size_t)e * 4 + 2];
  const float e1z = eattr[(size_t)e * 4 + 3];
  const int h0 = (threadIdx.x & 1) * 4;
#pragma unroll
  for (int h = h0; h < h0 + 4; ++h) {
    float acc = 0.f;
#pragma unroll
    for (int c = 0; c < ALPHA_C; ++c) {
      const int flat = h * ALPHA_C + c;
      float a;
      if (flat < NS) {
        float s = ssrc[(size_t)src * 64 + flat] + sdst[(size_t)dst * 64 + flat];
        a = w[flat] * s * e0;
      } else {
        const int k = flat - NS;
        float vx = vsrc[(size_t)src * 48 + k * 3 + 0] + vdst[(size_t)dst * 48 + k * 3 + 0];
        float vy = vsrc[(size_t)src * 48 + k * 3 + 1] + vdst[(size_t)dst * 48 + k * 3 + 1];
        float vz = vsrc[(size_t)src * 48 + k * 3 + 2] + vdst[(size_t)dst * 48 + k * 3 + 2];
        a = w[144 + k] * (vx * e1x + vy * e1y + vz * e1z) * INV_S3;
      }
      float sl = a * (0.2f + 0.8f / (1.f + expf(-a)));      // smooth_leaky
      acc += sl * alpha_dot[h * ALPHA_C + c];
    }
    alpha_out[(size_t)e * NHEAD + h] = acc;
    atomicMax(&amax_enc[(size_t)dst * NHEAD + h], fenc(acc));
  }
}

// ============================ softmax helpers ============================
__global__ void k_amax_fix(const unsigned* __restrict__ amax_enc,
                           float* __restrict__ amaxf, int N) {
  int idx = blockIdx.x * blockDim.x + threadIdx.x;
  if (idx >= N * NHEAD) return;
  float x = fdec(amax_enc[idx]);
  amaxf[idx] = isfinite(x) ? x : 0.f;
}

__global__ void k_den(const float* __restrict__ alpha, const float* __restrict__ amaxf,
                      const int* __restrict__ edst, float* __restrict__ den, int E) {
  long idx = (long)blockIdx.x * blockDim.x + threadIdx.x;
  if (idx >= (long)E * NHEAD) return;
  int e = (int)(idx >> 3), h = (int)(idx & 7);
  int dst = edst[e];
  float ex = expf(alpha[idx] - amaxf[(size_t)dst * NHEAD + h]);
  atomicAdd(&den[(size_t)dst * NHEAD + h], ex);
}

// ============================ edge pass 2: weighted scatter ============================
__global__ void __launch_bounds__(128)
k_edge_scatter(const float* __restrict__ xs, const _Float16* __restrict__ wpack,
               const float* __restrict__ b1, const float* __restrict__ b2,
               const int* __restrict__ esrc, const int* __restrict__ edst,
               const float* __restrict__ eattr,
               const float* __restrict__ ssrc, const float* __restrict__ sdst,
               const float* __restrict__ vsrc, const float* __restrict__ vdst,
               const float* __restrict__ alpha, const float* __restrict__ amaxf,
               const float* __restrict__ den, float* __restrict__ agg, int E) {
  __shared__ WaveScratch SW[4];
  const int wave = threadIdx.x >> 5, lane = threadIdx.x & 31;
  radial_mlp(SW[wave], blockIdx.x * 64 + wave * 16, lane, E, xs, wpack, b1, b2);
  __syncthreads();

  const int el = threadIdx.x >> 1;
  const int e  = blockIdx.x * 64 + el;
  if (e >= E) return;
  const float* w = SW[el >> 4].wbuf + (el & 15) * NW;
  const int src = esrc[e], dst = edst[e];
  const float e0  = eattr[(size_t)e * 4 + 0];
  const float e1x = eattr[(size_t)e * 4 + 1];
  const float e1y = eattr[(size_t)e * 4 + 2];
  const float e1z = eattr[(size_t)e * 4 + 3];
  const int h0 = (threadIdx.x & 1) * 4;
#pragma unroll
  for (int h = h0; h < h0 + 4; ++h) {
    const float attn = expf(alpha[(size_t)e * NHEAD + h] - amaxf[(size_t)dst * NHEAD + h])
                       / (den[(size_t)dst * NHEAD + h] + 1e-9f);
    float* arow = agg + (size_t)dst * AGG_C + h * 46;
    // scalar part of value = raw a_feat
#pragma unroll
    for (int c = 0; c < ALPHA_C; ++c) {
      const int flat = h * ALPHA_C + c;
      float a;
      if (flat < NS) {
        float s = ssrc[(size_t)src * 64 + flat] + sdst[(size_t)dst * 64 + flat];
        a = w[flat] * s * e0;
      } else {
        const int k = flat - NS;
        float vx = vsrc[(size_t)src * 48 + k * 3 + 0] + vdst[(size_t)dst * 48 + k * 3 + 0];
        float vy = vsrc[(size_t)src * 48 + k * 3 + 1] + vdst[(size_t)dst * 48 + k * 3 + 1];
        float vz = vsrc[(size_t)src * 48 + k * 3 + 2] + vdst[(size_t)dst * 48 + k * 3 + 2];
        a = w[144 + k] * (vx * e1x + vy * e1y + vz * e1z) * INV_S3;
      }
      atomicAdd(arow + c, attn * a);
    }
    // vector part: 12 channels x 3 components
#pragma unroll
    for (int j = 0; j < VEC_C; ++j) {
      const int ch = h * VEC_C + j;
      float ox, oy, oz;
      if (ch < 64) {
        float s = ssrc[(size_t)src * 64 + ch] + sdst[(size_t)dst * 64 + ch];
        float t = w[64 + ch] * s;
        ox = t * e1x; oy = t * e1y; oz = t * e1z;
      } else if (ch < 80) {
        const int k = ch - 64;
        float vx = vsrc[(size_t)src * 48 + k * 3 + 0] + vdst[(size_t)dst * 48 + k * 3 + 0];
        float vy = vsrc[(size_t)src * 48 + k * 3 + 1] + vdst[(size_t)dst * 48 + k * 3 + 1];
        float vz = vsrc[(size_t)src * 48 + k * 3 + 2] + vdst[(size_t)dst * 48 + k * 3 + 2];
        float t = w[128 + k] * e0;
        ox = t * vx; oy = t * vy; oz = t * vz;
      } else {
        const int k = ch - 80;
        float vx = vsrc[(size_t)src * 48 + k * 3 + 0] + vdst[(size_t)dst * 48 + k * 3 + 0];
        float vy = vsrc[(size_t)src * 48 + k * 3 + 1] + vdst[(size_t)dst * 48 + k * 3 + 1];
        float vz = vsrc[(size_t)src * 48 + k * 3 + 2] + vdst[(size_t)dst * 48 + k * 3 + 2];
        float t = w[160 + k] * INV_S2;
        ox = t * (vy * e1z - vz * e1y);
        oy = t * (vz * e1x - vx * e1z);
        oz = t * (vx * e1y - vy * e1x);
      }
      atomicAdd(arow + 10 + j * 3 + 0, attn * ox);
      atomicAdd(arow + 10 + j * 3 + 1, attn * oy);
      atomicAdd(arow + 10 + j * 3 + 2, attn * oz);
    }
  }
}

// ============================ output projections ============================
__global__ void __launch_bounds__(256)
k_out(const float* __restrict__ agg, const float* __restrict__ proj_ws,
      const float* __restrict__ proj_bs, const float* __restrict__ proj_wv,
      float* __restrict__ out, int N) {
  __shared__ float sws[80 * 64];
  __shared__ float swv[96 * 16];
  for (int i = threadIdx.x; i < 80 * 64; i += 256) sws[i] = proj_ws[i];
  for (int i = threadIdx.x; i < 96 * 16; i += 256) swv[i] = proj_wv[i];
  __syncthreads();
  long idx = (long)blockIdx.x * blockDim.x + threadIdx.x;
  if (idx >= (long)N * D_OUT) return;
  int n = (int)(idx / D_OUT), oc = (int)(idx % D_OUT);
  const float* arow = agg + (size_t)n * AGG_C;
  float acc;
  if (oc < 64) {
    acc = proj_bs[oc];
#pragma unroll
    for (int hh = 0; hh < NHEAD; ++hh)
#pragma unroll
      for (int c = 0; c < ALPHA_C; ++c)
        acc += arow[hh * 46 + c] * sws[(hh * ALPHA_C + c) * 64 + oc];
  } else {
    const int j = oc - 64, ev = j / 3, d = j % 3;
    acc = 0.f;
#pragma unroll
    for (int hh = 0; hh < NHEAD; ++hh)
#pragma unroll
      for (int c = 0; c < VEC_C; ++c)
        acc += arow[hh * 46 + 10 + c * 3 + d] * swv[(hh * VEC_C + c) * 16 + ev];
  }
  out[idx] = acc;
}

// ============================ launcher ============================
extern "C" void kernel_launch(void* const* d_in, const int* in_sizes, int n_in,
                              void* d_out, int out_size, void* d_ws, size_t ws_size,
                              hipStream_t stream) {
  const float* node_input   = (const float*)d_in[0];
  const int*   edge_src     = (const int*)d_in[1];
  const int*   edge_dst     = (const int*)d_in[2];
  const float* edge_attr    = (const float*)d_in[3];
  const float* edge_scalars = (const float*)d_in[4];
  const float* w_src_s      = (const float*)d_in[5];
  const float* w_src_v      = (const float*)d_in[6];
  const float* w_dst_s      = (const float*)d_in[7];
  const float* w_dst_v      = (const float*)d_in[8];
  const float* rad_w1       = (const float*)d_in[9];
  const float* rad_b1       = (const float*)d_in[10];
  const float* rad_w2       = (const float*)d_in[11];
  const float* rad_b2       = (const float*)d_in[12];
  const float* alpha_dot    = (const float*)d_in[13];
  const float* proj_ws      = (const float*)d_in[14];
  const float* proj_bs      = (const float*)d_in[15];
  const float* proj_wv      = (const float*)d_in[16];

  const int N = in_sizes[0] / D_IN;
  const int E = in_sizes[1];

  float* ws = (float*)d_ws;
  size_t o = 0;
  float*    ssrc = ws + o;  o += (size_t)N * 64;
  float*    sdst = ws + o;  o += (size_t)N * 64;
  float*    vsrc = ws + o;  o += (size_t)N * 48;
  float*    vdst = ws + o;  o += (size_t)N * 48;
  float*    alph = ws + o;  o += (size_t)E * NHEAD;
  unsigned* amen = (unsigned*)(ws + o); o += (size_t)N * NHEAD;
  float*    amax = ws + o;  o += (size_t)N * NHEAD;
  float*    den  = ws + o;  o += (size_t)N * NHEAD;
  float*    agg  = ws + o;  o += (size_t)N * AGG_C;
  o = (o + 15) & ~(size_t)15;                       // 32-byte align the f16 pack
  _Float16* wpack = (_Float16*)(ws + o);            // NFRAG*512 halves = 38912 B

  k_pack<<<(NFRAG * 512 + 255) / 256, 256, 0, stream>>>(rad_w1, rad_w2, wpack);
  {
    long tot = (long)N * (NHEAD + NHEAD + AGG_C);
    k_init<<<(int)((tot + 255) / 256), 256, 0, stream>>>(amen, den, agg, N);
  }
  k_node<<<(N + 31) / 32, 256, 0, stream>>>(node_input, w_src_s, w_src_v, w_dst_s, w_dst_v,
                                            ssrc, sdst, vsrc, vdst, N);
  k_edge_alpha<<<(E + 63) / 64, 128, 0, stream>>>(edge_scalars, wpack, rad_b1, rad_b2,
                                                  edge_src, edge_dst, edge_attr,
                                                  ssrc, sdst, vsrc, vdst, alpha_dot,
                                                  alph, amen, E);
  k_amax_fix<<<(N * NHEAD + 255) / 256, 256, 0, stream>>>(amen, amax, N);
  {
    long tot = (long)E * NHEAD;
    k_den<<<(int)((tot + 255) / 256), 256, 0, stream>>>(alph, amax, edge_dst, den, E);
  }
  k_edge_scatter<<<(E + 63) / 64, 128, 0, stream>>>(edge_scalars, wpack, rad_b1, rad_b2,
                                                    edge_src, edge_dst, edge_attr,
                                                    ssrc, sdst, vsrc, vdst,
                                                    alph, amax, den, agg, E);
  {
    long tot = (long)N * D_OUT;
    k_out<<<(int)((tot + 255) / 256), 256, 0, stream>>>(agg, proj_ws, proj_bs, proj_wv,
                                                        (float*)d_out, N);
  }
}
